// LAtt_45380624449723
// MI455X (gfx1250) — matrix operand order repaired and checked by reading, not compiled, not run on hardware
//
#include <hip/hip_runtime.h>
#include <hip/hip_bf16.h>
#include <stdint.h>

typedef __bf16 bf16;
typedef bf16  v16bf __attribute__((ext_vector_type(16)));
typedef float v8f   __attribute__((ext_vector_type(8)));
typedef unsigned int u32x4 __attribute__((ext_vector_type(4)));
typedef int i32x4 __attribute__((ext_vector_type(4)));
typedef int i32x8 __attribute__((ext_vector_type(8)));

constexpr int NB    = 16;        // batch
constexpr int NC    = 64;        // channels in/out
constexpr int NHID  = 128;       // heads*dim_head
constexpr int NHEAD = 4;
constexpr int NDH   = 32;
constexpr int NTOK  = 16384;     // 128*128
constexpr int CHUNK = 128;       // pixels per block
constexpr int S2    = 136;      // padded stride for row-major [row][px] buffers
constexpr int SQT   = 136;      // padded stride for transposed [px][hid] buffers
constexpr float EPS = 1e-5f;
constexpr int CN    = NC * NTOK; // 1,048,576 per batch

// workspace layout (float offsets)
constexpr int WS_GSUM1  = 0;
constexpr int WS_GSS1   = 16;
constexpr int WS_MU1    = 32;
constexpr int WS_RS1    = 48;
constexpr int WS_GSUM2  = 64;
constexpr int WS_GSS2   = 80;
constexpr int WS_MU2    = 96;
constexpr int WS_RS2    = 112;
constexpr int WS_SUMEXP = 128;                 // 16*128
constexpr int WS_CTX    = 128 + NB * NHID;     // 2176, 16*4*32*32
constexpr int WS_PROJ   = WS_CTX + NB * NHEAD * NDH * NDH;   // 67712
constexpr int WS_ZERO   = WS_PROJ;             // zero everything before proj

// ---- TDM: async 2D tile load Global -> LDS (CDNA5 ISA)
// D# group0: [1:0]=count=1, [63:32]=lds_addr, [120:64]=global_addr, [127:126]=2
// D# group1: [17:16]=data_size(2->4B), [79:48]=tensor_dim0, [111:80]=tensor_dim1,
//            [127:112]=tile_dim0, [143:128]=tile_dim1, [207:160]=dim0_stride
__device__ inline void tdm_load_2d(uint32_t lds_off, const void* gptr,
                                   uint32_t tile_d0, uint32_t tile_d1,
                                   uint32_t tensor_d0, uint32_t tensor_d1,
                                   uint32_t stride_e) {
  uint64_t ga = (uint64_t)(uintptr_t)gptr;
  u32x4 g0;
  g0[0] = 1u;                                              // count=1, user mode
  g0[1] = lds_off;                                         // LDS byte address
  g0[2] = (uint32_t)ga;                                    // global addr lo
  g0[3] = (uint32_t)((ga >> 32) & 0x1FFFFFFu) | (2u << 30);// addr hi | type=2
  i32x8 g1;
  g1[0] = (int)(2u << 16);                                 // data_size = 4B
  g1[1] = (int)((tensor_d0 & 0xFFFFu) << 16);              // tensor_dim0 lo16
  g1[2] = (int)((tensor_d0 >> 16) | ((tensor_d1 & 0xFFFFu) << 16));
  g1[3] = (int)((tensor_d1 >> 16) | (tile_d0 << 16));      // tile_dim0
  g1[4] = (int)(tile_d1 & 0xFFFFu);                        // tile_dim1 (2D)
  g1[5] = (int)stride_e;                                   // dim0 stride lo32
  g1[6] = 0;
  g1[7] = 0;
  i32x4 z4 = {0, 0, 0, 0};
#if defined(__clang_major__) && __clang_major__ >= 23
  i32x8 z8 = {0, 0, 0, 0, 0, 0, 0, 0};
  __builtin_amdgcn_tensor_load_to_lds(g0, g1, z4, z4, z8, 0);
#else
  __builtin_amdgcn_tensor_load_to_lds(g0, g1, z4, z4, 0);
#endif
}

// ---- WMMA tile loaders (contiguous k order: k = 16*(lane/16) + s, same in A and B) ----------

// A: 16x32 bf16 from row-major [M][stride] LDS (contiguous in K).
__device__ inline v16bf ldsA(const bf16* p, int stride, int m0, int k0, int lane) {
  const bf16* r = p + (m0 + (lane & 15)) * stride + k0 + ((lane & 16) ? 16 : 0);
  v16bf a;
#pragma unroll
  for (int i = 0; i < 8; ++i) a[i] = r[i];
#pragma unroll
  for (int i = 0; i < 8; ++i) a[8 + i] = r[8 + i];
  return a;
}

// B: 32x16 bf16 from row-major [K][stride] LDS (strided; used only where hoisted).
__device__ inline v16bf ldsB(const bf16* p, int stride, int k0, int n0, int lane) {
  const bf16* c0 = p + (k0 + ((lane & 16) ? 16 : 0)) * stride + n0 + (lane & 15);
  v16bf b;
#pragma unroll
  for (int i = 0; i < 16; ++i) b[i] = c0[i * stride];
  return b;
}

// B from N-major storage [N][stride] with B(k,n)=p[n*stride+k] (contiguous in K).
__device__ inline v16bf ldsBt(const bf16* p, int stride, int k0, int n0, int lane) {
  const bf16* r = p + (n0 + (lane & 15)) * stride + k0 + ((lane & 16) ? 16 : 0);
  v16bf b;
#pragma unroll
  for (int i = 0; i < 16; ++i) b[i] = r[i];
  return b;
}

// A from f32 ctx stored as [K=d][M=e] (32x32 per head): A(m,k) = p[k*32+m]
__device__ inline v16bf ldsAt_ctx(const float* p, int m0, int lane) {
  int m = m0 + (lane & 15);
  int kh = (lane & 16) ? 16 : 0;
  v16bf a;
#pragma unroll
  for (int i = 0; i < 8; ++i) a[i] = (bf16)p[(kh + i) * 32 + m];
#pragma unroll
  for (int i = 0; i < 8; ++i) a[8 + i] = (bf16)p[(kh + 8 + i) * 32 + m];
  return a;
}

__device__ inline v8f wmma_bf16(v16bf a, v16bf b, v8f c) {
  return __builtin_amdgcn_wmma_f32_16x16x32_bf16(false, a, false, b, (short)0, c,
                                                 false, false);
}

__device__ inline void cvt4(bf16* dst, float4 v) {
  dst[0] = (bf16)v.x; dst[1] = (bf16)v.y; dst[2] = (bf16)v.z; dst[3] = (bf16)v.w;
}

// normalize f32 x-tile staged in LDS -> bf16 row-major [c][px]
__device__ inline void norm_from_lds(bf16* sX, const float* sXF, const float* gw,
                                     const float* gb, float mu, float rs, int tid) {
  for (int i = tid; i < NC * CHUNK / 4; i += 256) {
    int c = i >> 5;             // CHUNK/4 == 32 groups per row
    int p4 = (i & 31) * 4;
    float4 v = *(const float4*)(sXF + c * CHUNK + p4);
    float sc = rs * gw[c];
    float of = gb[c] - mu * sc;
    bf16* dst = sX + c * S2 + p4;
    dst[0] = (bf16)(v.x * sc + of); dst[1] = (bf16)(v.y * sc + of);
    dst[2] = (bf16)(v.z * sc + of); dst[3] = (bf16)(v.w * sc + of);
  }
}

__device__ inline void prefetch_next(const float* xb, int tid, int last) {
  if (!last) {
    // next chunk: 64 rows x 512B; one 32-float segment per thread covers it
    int c = tid >> 2, seg = tid & 3;
    __builtin_prefetch(xb + CHUNK + (size_t)c * NTOK + seg * 32, 0, 0);
  }
}

// ---- kernels -------------------------------------------------------------

__global__ void k_zero(float* ws) {
  int i = blockIdx.x * blockDim.x + threadIdx.x;
  if (i < WS_ZERO) ws[i] = 0.f;
}

__global__ void __launch_bounds__(256) k_stats1(const float* __restrict__ x, float* ws) {
  int b = blockIdx.x, part = blockIdx.y;  // gridDim.y = 32
  const float4* xb = (const float4*)(x + (size_t)b * CN + (size_t)part * (CN / 32));
  float s = 0.f, ss = 0.f;
  for (int i = threadIdx.x; i < CN / 32 / 4; i += 256) {
    float4 v = xb[i];
    s  += v.x + v.y + v.z + v.w;
    ss += v.x * v.x + v.y * v.y + v.z * v.z + v.w * v.w;
  }
  __shared__ float r0[256], r1[256];
  r0[threadIdx.x] = s; r1[threadIdx.x] = ss;
  __syncthreads();
  for (int o = 128; o > 0; o >>= 1) {
    if (threadIdx.x < o) { r0[threadIdx.x] += r0[threadIdx.x + o];
                           r1[threadIdx.x] += r1[threadIdx.x + o]; }
    __syncthreads();
  }
  if (threadIdx.x == 0) {
    atomicAdd(&ws[WS_GSUM1 + b], r0[0]);
    atomicAdd(&ws[WS_GSS1 + b], r1[0]);
  }
}

__global__ void k_finstats(float* ws, int so, int qo, int mo, int ro) {
  int b = threadIdx.x;
  if (b < NB) {
    float m = ws[so + b] / (float)CN;
    float var = ws[qo + b] / (float)CN - m * m;
    ws[mo + b] = m;
    ws[ro + b] = rsqrtf(var + EPS);
  }
}

// pass 1: k,v = W_kv @ xn ; accumulate sum(e^k) and ctx_raw = e^k . v^T
__global__ void __launch_bounds__(256) k_ctx(const float* __restrict__ x,
                                             const float* __restrict__ gw,
                                             const float* __restrict__ gb,
                                             const float* __restrict__ qkvw,
                                             float* ws) {
  __shared__ __align__(16) bf16 sW[256 * 64];    // Wk rows 0-127, Wv rows 128-255
  __shared__ __align__(16) bf16 sX[NC * S2];     // normalized x chunk [c][px]
  __shared__ __align__(16) bf16 sEV[256 * S2];   // [row][px]: e^k 0-127, v/n 128-255
  int b = blockIdx.x;
  int n0g = blockIdx.y * CHUNK;
  int tid = threadIdx.x, lane = tid & 31, wv = tid >> 5;
  float mu = ws[WS_MU1 + b], rs = ws[WS_RS1 + b];
  const float* xb = x + (size_t)b * CN + n0g;
  float* sXF = (float*)sEV;   // f32 staging tile overlaps sEV (dead until later)

  if (wv == 0)                 // one TDM issue per block (uniform branch)
    tdm_load_2d((uint32_t)(uintptr_t)(void*)sXF, xb, CHUNK, NC, NTOK, NC, NTOK);
  prefetch_next(xb, tid, blockIdx.y + 1 == gridDim.y);
  for (int i = tid; i < 256 * 64 / 4; i += 256)
    cvt4(sW + i * 4, *(const float4*)(qkvw + 128 * 64 + i * 4));
  __builtin_amdgcn_s_wait_tensorcnt(0);
  __syncthreads();
  norm_from_lds(sX, sXF, gw, gb, mu, rs, tid);
  __syncthreads();

  // k/v projection: 16 mtiles x 8 ntiles (nt fixed per wave -> B hoisted), K=64
  for (int t = wv; t < 128; t += 8) {
    int m0 = (t >> 3) * 16, n0 = (t & 7) * 16;
    v8f acc = {};
#pragma unroll
    for (int ks = 0; ks < 2; ++ks)
      acc = wmma_bf16(ldsA(sW, 64, m0, ks * 32, lane),
                      ldsB(sX, S2, ks * 32, n0, lane), acc);
    bool isk = (m0 < 128);
#pragma unroll
    for (int r = 0; r < 8; ++r) {
      int row = m0 + r + ((lane & 16) ? 8 : 0);
      int col = n0 + (lane & 15);
      float v = acc[r];
      v = isk ? __expf(v) : v * (1.0f / (float)NTOK);
      sEV[row * S2 + col] = (bf16)v;
    }
  }
  __syncthreads();

  // partial sum(e^k) per row d (contiguous reads)
  if (tid < 128) {
    float s = 0.f;
#pragma unroll 4
    for (int j = 0; j < CHUNK; ++j) s += (float)sEV[tid * S2 + j];
    atomicAdd(&ws[WS_SUMEXP + b * NHID + tid], s);
  }

  // ctx_raw[h][d][e] += e^k[d,:] . v[e,:]  (A=e^k MxK over px, B=v^T; both contig)
  for (int t = wv; t < 16; t += 8) {
    int h = t >> 2, m0 = ((t >> 1) & 1) * 16, n0 = (t & 1) * 16;
    const bf16* ek = sEV + (h * 32) * S2;
    const bf16* vv = sEV + (128 + h * 32) * S2;
    v8f acc = {};
#pragma unroll
    for (int ks = 0; ks < 4; ++ks)
      acc = wmma_bf16(ldsA(ek, S2, m0, ks * 32, lane),
                      ldsBt(vv, S2, ks * 32, n0, lane), acc);
    float* ctx = ws + WS_CTX + (size_t)(b * NHEAD + h) * 32 * 32;
#pragma unroll
    for (int r = 0; r < 8; ++r) {
      int d = m0 + r + ((lane & 16) ? 8 : 0);
      int e = n0 + (lane & 15);
      atomicAdd(&ctx[d * 32 + e], acc[r]);
    }
  }
}

__global__ void k_ctxfin(float* ws) {
  int i = blockIdx.x * blockDim.x + threadIdx.x;
  if (i < NB * NHEAD * 32 * 32) {
    int d = (i >> 5) & 31, h = (i >> 10) & 3, b = i >> 12;
    float se = ws[WS_SUMEXP + b * NHID + h * 32 + d];
    ws[WS_CTX + i] = ws[WS_CTX + i] * rsqrtf((float)NDH) / se;  // fold q scale
  }
}

// pass 2: q proj + softmax_d, hid = ctx^T @ q, proj = Wo @ hid + b; GN2 partials
__global__ void __launch_bounds__(256) k_out(const float* __restrict__ x,
                                             const float* __restrict__ gw,
                                             const float* __restrict__ gb,
                                             const float* __restrict__ qkvw,
                                             const float* __restrict__ outw,
                                             const float* __restrict__ outb,
                                             float* ws) {
  __shared__ __align__(16) bf16 sWq[128 * 64];
  __shared__ __align__(16) bf16 sWo[64 * 128];
  __shared__ __align__(16) bf16 sX[NC * S2];       // [c][px]
  __shared__ __align__(16) bf16 sQ[CHUNK * SQT];   // transposed [px][hid]
  __shared__ __align__(16) bf16 sH[CHUNK * SQT];   // transposed [px][hid]
  __shared__ __align__(16) float sCtx[NHEAD * 32 * 32];
  __shared__ float sRed[512];
  int b = blockIdx.x, n0g = blockIdx.y * CHUNK;
  int tid = threadIdx.x, lane = tid & 31, wv = tid >> 5;
  int lh8 = (lane & 16) ? 8 : 0;
  float mu = ws[WS_MU1 + b], rs = ws[WS_RS1 + b];
  const float* xb = x + (size_t)b * CN + n0g;
  float* sXF = (float*)sH;    // f32 staging tile overlaps sH (dead until later)

  if (wv == 0)                 // one TDM issue per block (uniform branch)
    tdm_load_2d((uint32_t)(uintptr_t)(void*)sXF, xb, CHUNK, NC, NTOK, NC, NTOK);
  prefetch_next(xb, tid, blockIdx.y + 1 == gridDim.y);
  for (int i = tid; i < 128 * 64 / 4; i += 256)
    cvt4(sWq + i * 4, *(const float4*)(qkvw + i * 4));
  for (int i = tid; i < 64 * 128 / 4; i += 256)
    cvt4(sWo + i * 4, *(const float4*)(outw + i * 4));
  for (int i = tid; i < NHEAD * 32 * 32 / 4; i += 256)
    *(float4*)(sCtx + i * 4) =
        *(const float4*)(ws + WS_CTX + (size_t)b * NHEAD * 32 * 32 + i * 4);
  __builtin_amdgcn_s_wait_tensorcnt(0);
  __syncthreads();
  norm_from_lds(sX, sXF, gw, gb, mu, rs, tid);
  __syncthreads();

  // q_pre = Wq @ xn : 8x8 tiles, K=64; store transposed -> contiguous b128
  for (int t = wv; t < 64; t += 8) {
    int m0 = (t >> 3) * 16, n0 = (t & 7) * 16;
    v8f acc = {};
#pragma unroll
    for (int ks = 0; ks < 2; ++ks)
      acc = wmma_bf16(ldsA(sWq, 64, m0, ks * 32, lane),
                      ldsB(sX, S2, ks * 32, n0, lane), acc);
    bf16* dst = sQ + (n0 + (lane & 15)) * SQT + m0 + lh8;
#pragma unroll
    for (int r = 0; r < 8; ++r) dst[r] = (bf16)acc[r];
  }
  __syncthreads();

  // softmax over d (32 contiguous bf16 per (head,px)); scale folded into ctx
  for (int i = tid; i < NHEAD * CHUNK; i += 256) {
    int h = i >> 7, col = i & (CHUNK - 1);
    bf16* qp = sQ + col * SQT + h * 32;
    float vals[32], m = -1e30f;
#pragma unroll
    for (int d = 0; d < 32; ++d) { vals[d] = (float)qp[d]; m = fmaxf(m, vals[d]); }
    float s = 0.f;
#pragma unroll
    for (int d = 0; d < 32; ++d) { vals[d] = __expf(vals[d] - m); s += vals[d]; }
    float inv = 1.f / s;
#pragma unroll
    for (int d = 0; d < 32; ++d) qp[d] = (bf16)(vals[d] * inv);
  }
  __syncthreads();

  // hid[e,n] = sum_d ctx[d,e] q[d,n] : per head 2x8 tiles, K=32; B contiguous
  for (int t = wv; t < 64; t += 8) {
    int h = t >> 4, m0 = ((t >> 3) & 1) * 16, n0 = (t & 7) * 16;
    v8f acc = {};
    acc = wmma_bf16(ldsAt_ctx(sCtx + h * 1024, m0, lane),
                    ldsBt(sQ + h * 32, SQT, 0, n0, lane), acc);
    bf16* dst = sH + (n0 + (lane & 15)) * SQT + h * 32 + m0 + lh8;
#pragma unroll
    for (int r = 0; r < 8; ++r) dst[r] = (bf16)acc[r];
  }
  __syncthreads();

  // proj = Wo @ hid + out_b : 4x8 tiles, K=128; GN2 partial stats
  float lsum = 0.f, lss = 0.f;
  float* proj = ws + WS_PROJ + (size_t)b * CN + n0g;
  for (int t = wv; t < 32; t += 8) {
    int m0 = (t >> 3) * 16, n0 = (t & 7) * 16;
    v8f acc = {};
#pragma unroll
    for (int ks = 0; ks < 4; ++ks)
      acc = wmma_bf16(ldsA(sWo, 128, m0, ks * 32, lane),
                      ldsBt(sH, SQT, ks * 32, n0, lane), acc);
#pragma unroll
    for (int r = 0; r < 8; ++r) {
      int c = m0 + r + lh8;
      int col = n0 + (lane & 15);
      float v = acc[r] + outb[c];
      proj[(size_t)c * NTOK + col] = v;
      lsum += v; lss += v * v;
    }
  }
  sRed[tid] = lsum; sRed[256 + tid] = lss;
  __syncthreads();
  for (int o = 128; o > 0; o >>= 1) {
    if (tid < o) { sRed[tid] += sRed[tid + o];
                   sRed[256 + tid] += sRed[256 + tid + o]; }
    __syncthreads();
  }
  if (tid == 0) {
    atomicAdd(&ws[WS_GSUM2 + b], sRed[0]);
    atomicAdd(&ws[WS_GSS2 + b], sRed[256]);
  }
}

__global__ void k_final(const float* __restrict__ x, const float* __restrict__ g2w,
                        const float* __restrict__ g2b, const float* __restrict__ ws,
                        float* __restrict__ out) {
  size_t i4 = (size_t)blockIdx.x * blockDim.x + threadIdx.x;
  if (i4 < (size_t)NB * CN / 4) {
    size_t i = i4 * 4;
    int b = (int)(i / CN);
    int c = (int)((i % CN) / NTOK);   // 4 consecutive elems share c (NTOK%4==0)
    float mu = ws[WS_MU2 + b], rs = ws[WS_RS2 + b];
    float sc = rs * g2w[c];
    float of = g2b[c] - mu * sc;
    float4 p = *(const float4*)(ws + WS_PROJ + i);
    float4 xv = *(const float4*)(x + i);
    float4 o;
    o.x = p.x * sc + of + xv.x;
    o.y = p.y * sc + of + xv.y;
    o.z = p.z * sc + of + xv.z;
    o.w = p.w * sc + of + xv.w;
    *(float4*)(out + i) = o;
  }
}

extern "C" void kernel_launch(void* const* d_in, const int* in_sizes, int n_in,
                              void* d_out, int out_size, void* d_ws, size_t ws_size,
                              hipStream_t stream) {
  const float* x     = (const float*)d_in[0];
  const float* gn1_w = (const float*)d_in[1];
  const float* gn1_b = (const float*)d_in[2];
  const float* qkv_w = (const float*)d_in[3];
  const float* out_w = (const float*)d_in[4];
  const float* out_b = (const float*)d_in[5];
  const float* gn2_w = (const float*)d_in[6];
  const float* gn2_b = (const float*)d_in[7];
  float* out = (float*)d_out;
  float* ws  = (float*)d_ws;

  k_zero<<<(WS_ZERO + 255) / 256, 256, 0, stream>>>(ws);
  k_stats1<<<dim3(NB, 32), 256, 0, stream>>>(x, ws);
  k_finstats<<<1, 32, 0, stream>>>(ws, WS_GSUM1, WS_GSS1, WS_MU1, WS_RS1);
  k_ctx<<<dim3(NB, NTOK / CHUNK), 256, 0, stream>>>(x, gn1_w, gn1_b, qkv_w, ws);
  k_ctxfin<<<(NB * NHEAD * 32 * 32 + 255) / 256, 256, 0, stream>>>(ws);
  k_out<<<dim3(NB, NTOK / CHUNK), 256, 0, stream>>>(x, gn1_w, gn1_b, qkv_w,
                                                    out_w, out_b, ws);
  k_finstats<<<1, 32, 0, stream>>>(ws, WS_GSUM2, WS_GSS2, WS_MU2, WS_RS2);
  k_final<<<(size_t)(NB * (size_t)CN / 4 + 255) / 256, 256, 0, stream>>>(
      x, gn2_w, gn2_b, ws, out);
}
